// Net_47132971107101
// MI455X (gfx1250) — matrix operand order repaired and checked by reading, not compiled, not run on hardware
//
#include <hip/hip_runtime.h>
#include <hip/hip_bf16.h>

typedef __attribute__((ext_vector_type(16))) _Float16 v16h;
typedef __attribute__((ext_vector_type(8)))  float    v8f;
typedef __attribute__((ext_vector_type(4)))  float    f32x4;

union V16H {
    v16h  v;
    f32x4 f4[2];
};

__device__ __forceinline__ float clamp1(float v) {
    return fminf(fmaxf(v, -1.0f), 1.0f);
}
// 2-bit unsigned activation quant: relu, clamp 6, round to {0,2,4,6} (RNE)
__device__ __forceinline__ float qrelu2(float v) {
    float y = fminf(fmaxf(v, 0.0f), 6.0f);
    return rintf(y * 0.5f) * 2.0f;
}

#define IMGS_PER_BLOCK 32
#define THREADS 256

__global__ __launch_bounds__(THREADS)
void qnet_fused_kernel(const float* __restrict__ x,
                       const float* __restrict__ w1,
                       const float* __restrict__ w2,
                       const float* __restrict__ wf,
                       float* __restrict__ out,
                       int Btot) {
    __shared__ float red[THREADS];
    __shared__ float w1q[4 * 9];        // [ch][3x3]
    __shared__ float w2q[4 * 4 * 9];    // [co][ci][3x3]
    // B matrix (wfq^T, K=224 padded, N=16 padded), pre-swizzled per lane (contiguous k order, same as A):
    // Bswz[kk][lane][sl] = B[K = 32*kk + (lane/16)*16 + sl][N = lane%16]
    __shared__ _Float16 Bswz[7][32][16] __attribute__((aligned(16)));
    // A matrix staging (2 tiles of 16 images), pre-swizzled per lane (contiguous k order, same as B):
    // koff = (lane/16)*16 + sl ; M = lane%16
    __shared__ _Float16 Aswz[2][7][32][16] __attribute__((aligned(16)));
    // h1: conv1+quant+pool result, values in {0,2,4,6} (exact in f16)
    __shared__ _Float16 h1s[IMGS_PER_BLOCK][4][14][14];

    const int tid = threadIdx.x;
    const int blk = blockIdx.x;
    const int img0 = blk * IMGS_PER_BLOCK;

    // ---------------- Phase 0: quant scales + quantized weights + B swizzle ----
    float m;
    // s_f = max|wf| over 10*196
    m = 0.0f;
    for (int i = tid; i < 1960; i += THREADS) m = fmaxf(m, fabsf(wf[i]));
    red[tid] = m; __syncthreads();
    for (int s = THREADS / 2; s > 0; s >>= 1) {
        if (tid < s) red[tid] = fmaxf(red[tid], red[tid + s]);
        __syncthreads();
    }
    const float sf = red[0]; __syncthreads();
    // s_1 = max|w1| over 36
    m = 0.0f;
    if (tid < 36) m = fabsf(w1[tid]);
    red[tid] = m; __syncthreads();
    for (int s = THREADS / 2; s > 0; s >>= 1) {
        if (tid < s) red[tid] = fmaxf(red[tid], red[tid + s]);
        __syncthreads();
    }
    const float s1 = red[0]; __syncthreads();
    // s_2 = max|w2| over 144
    m = 0.0f;
    if (tid < 144) m = fabsf(w2[tid]);
    red[tid] = m; __syncthreads();
    for (int s = THREADS / 2; s > 0; s >>= 1) {
        if (tid < s) red[tid] = fmaxf(red[tid], red[tid + s]);
        __syncthreads();
    }
    const float s2 = red[0]; __syncthreads();

    if (tid < 36)  w1q[tid] = clamp1(rintf(w1[tid] / s1)) * s1;
    if (tid < 144) w2q[tid] = clamp1(rintf(w2[tid] / s2)) * s2;

    // Build swizzled, quantized, padded B matrix in LDS
    for (int idx = tid; idx < 7 * 32 * 16; idx += THREADS) {
        const int kk   = idx >> 9;           // /512
        const int rem  = idx & 511;
        const int lane = rem >> 4;
        const int sl   = rem & 15;
        const int n    = lane & 15;
        const int K    = kk * 32 + ((lane >> 4) << 4) + sl;
        float v = 0.0f;
        if (n < 10 && K < 196)
            v = clamp1(rintf(wf[n * 196 + K] / sf)) * sf;
        Bswz[kk][lane][sl] = (_Float16)v;
    }
    // Zero A staging (provides K padding 196..223)
    {
        unsigned int* az = (unsigned int*)&Aswz[0][0][0][0];
        for (int i = tid; i < 2 * 7 * 32 * 16 / 2; i += THREADS) az[i] = 0u;
    }
    __syncthreads();

    // ---------------- Phase 1: conv1 (1->4) + qrelu2 + maxpool2x2 -> h1s ------
    for (int idx = tid; idx < IMGS_PER_BLOCK * 196; idx += THREADS) {
        const int img = idx / 196;
        const int p   = idx % 196;
        const int py  = p / 14, px = p % 14;
        const int imgg = img0 + img;
        const bool inb = (imgg < Btot);
        const float* xs = x + (size_t)imgg * 784;

        float acc[4][4];
        #pragma unroll
        for (int c = 0; c < 4; ++c)
            #pragma unroll
            for (int w = 0; w < 4; ++w) acc[c][w] = 0.0f;

        #pragma unroll
        for (int dy = 0; dy < 2; ++dy)
        #pragma unroll
        for (int dx = 0; dx < 2; ++dx) {
            const int oy = 2 * py + dy, ox = 2 * px + dx;
            #pragma unroll
            for (int ky = 0; ky < 3; ++ky) {
                const int iy = oy - 1 + ky;
                if (iy < 0 || iy >= 28) continue;
                #pragma unroll
                for (int kx = 0; kx < 3; ++kx) {
                    const int ix = ox - 1 + kx;
                    if (ix < 0 || ix >= 28) continue;
                    const float xv = inb ? xs[iy * 28 + ix] : 0.0f;
                    const int wi = ky * 3 + kx;
                    #pragma unroll
                    for (int c = 0; c < 4; ++c)
                        acc[c][dy * 2 + dx] += xv * w1q[c * 9 + wi];
                }
            }
        }
        #pragma unroll
        for (int c = 0; c < 4; ++c) {
            float q0 = qrelu2(acc[c][0]), q1 = qrelu2(acc[c][1]);
            float q2 = qrelu2(acc[c][2]), q3 = qrelu2(acc[c][3]);
            h1s[img][c][py][px] = (_Float16)fmaxf(fmaxf(q0, q1), fmaxf(q2, q3));
        }
    }
    __syncthreads();

    // ---------------- Phase 2: conv2 (4->4) + qrelu2 + maxpool -> Aswz --------
    for (int idx = tid; idx < IMGS_PER_BLOCK * 49; idx += THREADS) {
        const int img = idx / 49;
        const int p   = idx % 49;
        const int py  = p / 7, px = p % 7;

        float acc[4][4];
        #pragma unroll
        for (int c = 0; c < 4; ++c)
            #pragma unroll
            for (int w = 0; w < 4; ++w) acc[c][w] = 0.0f;

        #pragma unroll
        for (int dy = 0; dy < 2; ++dy)
        #pragma unroll
        for (int dx = 0; dx < 2; ++dx) {
            const int oy = 2 * py + dy, ox = 2 * px + dx;   // on 14x14 grid
            #pragma unroll
            for (int ci = 0; ci < 4; ++ci)
            #pragma unroll
            for (int ky = 0; ky < 3; ++ky) {
                const int iy = oy - 1 + ky;
                if (iy < 0 || iy >= 14) continue;
                #pragma unroll
                for (int kx = 0; kx < 3; ++kx) {
                    const int ix = ox - 1 + kx;
                    if (ix < 0 || ix >= 14) continue;
                    const float hv = (float)h1s[img][ci][iy][ix];
                    const int wi = ci * 9 + ky * 3 + kx;
                    #pragma unroll
                    for (int co = 0; co < 4; ++co)
                        acc[co][dy * 2 + dx] += hv * w2q[co * 36 + wi];
                }
            }
        }
        const int tile = img >> 4;
        const int mrow = img & 15;
        #pragma unroll
        for (int co = 0; co < 4; ++co) {
            float q0 = qrelu2(acc[co][0]), q1 = qrelu2(acc[co][1]);
            float q2 = qrelu2(acc[co][2]), q3 = qrelu2(acc[co][3]);
            const float val = fmaxf(fmaxf(q0, q1), fmaxf(q2, q3));
            const int k    = co * 49 + p;         // flatten index 0..195
            const int kk   = k >> 5;
            const int koff = k & 31;
            // inverse of A-layout map: koff = (lane/16)*16 + sl
            const int sl   = koff & 15;
            const int lane = ((koff >> 4) << 4) + mrow;
            Aswz[tile][kk][lane][sl] = (_Float16)val;
        }
    }
    __syncthreads();

    // ---------------- Phase 3: FC via WMMA, waves 0..1 -> 2 tiles of 16 imgs --
    const int wave = tid >> 5;
    const int lane = tid & 31;
    if (wave < 2) {                 // wave-uniform: EXEC stays all-1s for WMMA
        v8f c = {};
        #pragma unroll
        for (int kk = 0; kk < 7; ++kk) {
            V16H a, b;
            const f32x4* ap = (const f32x4*)&Aswz[wave][kk][lane][0];
            a.f4[0] = ap[0]; a.f4[1] = ap[1];
            const f32x4* bp = (const f32x4*)&Bswz[kk][lane][0];
            b.f4[0] = bp[0]; b.f4[1] = bp[1];
            c = __builtin_amdgcn_wmma_f32_16x16x32_f16(
                    false, a.v, false, b.v, (short)0, c, false, false);
        }
        const int n = lane & 15;
        if (n < 10) {
            #pragma unroll
            for (int r = 0; r < 8; ++r) {
                const int mrow = r + ((lane >> 4) << 3);
                const int imgg = img0 + wave * 16 + mrow;
                if (imgg < Btot) out[(size_t)imgg * 10 + n] = c[r];
            }
        }
    }
}

extern "C" void kernel_launch(void* const* d_in, const int* in_sizes, int n_in,
                              void* d_out, int out_size, void* d_ws, size_t ws_size,
                              hipStream_t stream) {
    const float* x  = (const float*)d_in[0];
    const float* w1 = (const float*)d_in[1];
    const float* w2 = (const float*)d_in[2];
    const float* wf = (const float*)d_in[3];
    float* out = (float*)d_out;

    const int Btot = in_sizes[0] / 784;          // 32768
    const int grid = (Btot + IMGS_PER_BLOCK - 1) / IMGS_PER_BLOCK;

    qnet_fused_kernel<<<grid, THREADS, 0, stream>>>(x, w1, w2, wf, out, Btot);
}
